// SelfAttention_35450660061483
// MI455X (gfx1250) — matrix operand order repaired and checked by reading, not compiled, not run on hardware
//
#include <hip/hip_runtime.h>
#include <hip/hip_bf16.h>

// MI455X / gfx1250, wave32. Matmuls on v_wmma_f32_16x16x32_bf16; hot-loop
// operand staging via the Tensor Data Mover (tensor_load_to_lds) with
// double-buffered LDS and s_wait_tensorcnt synchronization.

typedef __bf16 bf16_t;
typedef __attribute__((ext_vector_type(16))) __bf16 v16bf;
typedef __attribute__((ext_vector_type(8)))  __bf16 v8bf;
typedef __attribute__((ext_vector_type(8)))  float  v8f;
typedef __attribute__((ext_vector_type(4)))  unsigned int v4u;
typedef __attribute__((ext_vector_type(8)))  int v8i;
typedef __attribute__((ext_vector_type(4)))  int v4i;

#define B_DIM 8
#define L_DIM 2048
#define H_DIM 256

__device__ __forceinline__ bf16_t f2bf(float f) {
  union { float f; unsigned u; } v; v.f = f;
  unsigned r = (v.u + 0x7FFFu + ((v.u >> 16) & 1u)) >> 16;   // round-nearest-even
  union { unsigned short s; bf16_t b; } o; o.s = (unsigned short)r;
  return o.b;
}

__device__ __forceinline__ v8f vzero8() {
  v8f z = {0.f, 0.f, 0.f, 0.f, 0.f, 0.f, 0.f, 0.f};
  return z;
}

__device__ __forceinline__ v8f wmma_bf16(v16bf a, v16bf b, v8f c) {
  return __builtin_amdgcn_wmma_f32_16x16x32_bf16(false, a, false, b, (short)0, c,
                                                 false, false);
}

// A fragment (16x32 of a row-major matrix). Lane m: K=[g*16,g*16+16).
__device__ __forceinline__ v16bf load_a_frag(const bf16_t* base, int ld, int lane) {
  const int row = lane & 15, grp = (lane >> 4) & 1;
  const bf16_t* p = base + row * ld + grp * 16;
  v8bf lo = *(const v8bf*)(p);
  v8bf hi = *(const v8bf*)(p + 8);
  v16bf a;
#pragma unroll
  for (int e = 0; e < 8; ++e) { a[e] = lo[e]; a[e + 8] = hi[e]; }
  return a;
}

// B fragment for A @ S^T: column n of B = row n of row-major S.
__device__ __forceinline__ v16bf load_bt_frag(const bf16_t* base, int ld, int lane) {
  const int col = lane & 15, grp = (lane >> 4) & 1;
  return *(const v16bf*)(base + col * ld + grp * 16);
}

// ---------------------------------------------------------------------------
// TDM: DMA a 2D bf16 tile (rows x row_elems, row stride stride_elems) from
// global memory into LDS at byte offset lds_addr. D# per ISA 08 §8.3/8.4.
// clang-23 toolchain: 6-arg builtin (g0, g1, g2, g3, g_extra, cpol).
// ---------------------------------------------------------------------------
__device__ __forceinline__ void tdm_load_2d(unsigned lds_addr, const void* gptr,
                                            unsigned rows, unsigned row_elems,
                                            unsigned stride_elems) {
  const unsigned long long ga = (unsigned long long)gptr;
  v4u g0;
  g0[0] = 1u;                                              // count=1 (valid), user mode
  g0[1] = lds_addr;                                        // LDS byte address
  g0[2] = (unsigned)(ga & 0xFFFFFFFFu);                    // global_addr[31:0]
  g0[3] = (unsigned)((ga >> 32) & 0x01FFFFFFu) | (2u << 30); // addr[56:32] | type=2
  v8i g1;
  g1[0] = (int)(1u << 16);                                 // data_size=1 (2 bytes)
  g1[1] = (int)((row_elems & 0xFFFFu) << 16);              // tensor_dim0[15:0]
  g1[2] = (int)((row_elems >> 16) | ((rows & 0xFFFFu) << 16));     // dim0 hi | dim1 lo
  g1[3] = (int)((rows >> 16) | ((row_elems & 0xFFFFu) << 16));     // dim1 hi | tile_dim0
  g1[4] = (int)(rows & 0xFFFFu);                           // tile_dim1 (tile_dim2=0)
  g1[5] = (int)stride_elems;                               // tensor_dim0_stride[31:0]
  g1[6] = 0;
  g1[7] = 0;
  const v4i z4 = {0, 0, 0, 0};
  const v8i z8 = {0, 0, 0, 0, 0, 0, 0, 0};
  __builtin_amdgcn_tensor_load_to_lds(g0, g1, z4, z4, z8, 0);
}

__device__ __forceinline__ unsigned lds_off(const void* p) {
  return (unsigned)(unsigned long long)p;   // generic LDS ptr: addr[31:0] = LDS offset
}

// ---------------------------------------------------------------------------
// K0: f32 -> bf16 conversion (grid-stride)
// ---------------------------------------------------------------------------
__global__ void cvt_f32_bf16(const float* __restrict__ in, bf16_t* __restrict__ out, int n) {
  int i = blockIdx.x * blockDim.x + threadIdx.x;
  int stride = gridDim.x * blockDim.x;
  for (; i < n; i += stride) out[i] = f2bf(in[i]);
}

// ---------------------------------------------------------------------------
// K1: Wh = g @ WP^T   ([B*L,256] x [256,256]); one 16x16 tile per wave.
// ---------------------------------------------------------------------------
__global__ void wh_gemm_kernel(const bf16_t* __restrict__ g_bf,
                               const bf16_t* __restrict__ wp_bf,
                               bf16_t* __restrict__ wh_bf) {
  const int lane = threadIdx.x & 31;
  const int gw = blockIdx.x * 8 + (threadIdx.x >> 5);
  const int tm = gw >> 4;
  const int tn = gw & 15;
  const bf16_t* A  = g_bf  + tm * 16 * H_DIM;
  const bf16_t* Bm = wp_bf + tn * 16 * H_DIM;
  v8f acc = vzero8();
#pragma unroll
  for (int k = 0; k < 8; ++k)
    acc = wmma_bf16(load_a_frag(A + k * 32, H_DIM, lane),
                    load_bt_frag(Bm + k * 32, H_DIM, lane), acc);
  const int col = lane & 15, rowo = (lane >> 4) * 8;
  bf16_t* D = wh_bf + tm * 16 * H_DIM + tn * 16;
#pragma unroll
  for (int v = 0; v < 8; ++v) D[(rowo + v) * H_DIM + col] = f2bf(acc[v]);
}

// ---------------------------------------------------------------------------
// K2: column softmax stats. All 8 waves of a block share the same Wh i-tile
// stream -> stage it once per step via TDM (double-buffered 16x256 bf16).
// ---------------------------------------------------------------------------
__global__ void colstats_kernel(const bf16_t* __restrict__ wh_bf,
                                const bf16_t* __restrict__ g_bf,
                                float* __restrict__ cmax,
                                float* __restrict__ csum) {
  __shared__ __align__(32) bf16_t lds_wh[2][16][H_DIM];   // 2 x 8KB
  const int tid = threadIdx.x, lane = tid & 31, w = tid >> 5;
  const int gw = blockIdx.x * 8 + w;          // 1024 waves, block shares b
  const int b = gw >> 7, jt = gw & 127;
  const int rowbase = b * L_DIM;

  const bf16_t* gB = g_bf + (rowbase + jt * 16) * H_DIM;
  v16bf bfr[8];                                // hoist B (g rows j), per-wave
#pragma unroll
  for (int k = 0; k < 8; ++k) bfr[k] = load_bt_frag(gB + k * 32, H_DIM, lane);

  if (w == 0)
    tdm_load_2d(lds_off(&lds_wh[0][0][0]), wh_bf + (size_t)rowbase * H_DIM,
                16, H_DIM, H_DIM);

  float m = -__builtin_inff(), ssum = 0.0f;
  for (int it = 0; it < L_DIM / 16; ++it) {
    const int cur = it & 1;
    if (w == 0) __builtin_amdgcn_s_wait_tensorcnt(0);
    __syncthreads();                           // buf[cur] ready; old reads done
    if (w == 0 && it + 1 < L_DIM / 16)         // prefetch next tile via TDM
      tdm_load_2d(lds_off(&lds_wh[cur ^ 1][0][0]),
                  wh_bf + (size_t)(rowbase + (it + 1) * 16) * H_DIM,
                  16, H_DIM, H_DIM);
    const bf16_t* A = &lds_wh[cur][0][0];
    v8f s = vzero8();
#pragma unroll
    for (int k = 0; k < 8; ++k)
      s = wmma_bf16(load_a_frag(A + k * 32, H_DIM, lane), bfr[k], s);
#pragma unroll
    for (int v = 0; v < 8; ++v) {              // online softmax accumulate
      float x = s[v];
      float mn = fmaxf(m, x);
      ssum = ssum * __expf(m - mn) + __expf(x - mn);
      m = mn;
    }
  }
  float m2 = __shfl_xor(m, 16, 32);
  float s2 = __shfl_xor(ssum, 16, 32);
  float mn = fmaxf(m, m2);
  float tot = ssum * __expf(m - mn) + s2 * __expf(m2 - mn);
  if (lane < 16) {
    cmax[rowbase + jt * 16 + lane] = mn;
    csum[rowbase + jt * 16 + lane] = tot;
  }
}

// ---------------------------------------------------------------------------
// K3: c = softmax_col(s) @ g with s recompute. The 32 g-rows per j-step are
// TDM-staged once per block (double buffer); s-tile B-fragments and the g^T
// scatter both read from LDS. P bounced through per-wave LDS to reach the
// A-operand layout.
// ---------------------------------------------------------------------------
__global__ void ct_kernel(const bf16_t* __restrict__ wh_bf,
                          const bf16_t* __restrict__ g_bf,
                          const float* __restrict__ g32,
                          const float* __restrict__ cmax,
                          const float* __restrict__ csum,
                          bf16_t* __restrict__ c_bf,
                          bf16_t* __restrict__ gc_bf) {
  __shared__ __align__(32) bf16_t lds_g[2][32][H_DIM];   // 2 x 16KB, TDM target
  __shared__ __align__(32) bf16_t gt[64][32];            // g^T tile [h][j]
  __shared__ __align__(32) bf16_t pshm[8][16][32];       // per-wave P tile [i][j]
  const int tid = threadIdx.x, lane = tid & 31, w = tid >> 5;
  const int bid = blockIdx.x;                            // 8*16*4 = 512 blocks
  const int b  = bid >> 6;
  const int ig = (bid >> 2) & 15;
  const int hg = bid & 3;
  const int ibase = (ig * 8 + w) * 16;
  const int hbase = hg * 64;
  const int rowbase = b * L_DIM;
  const int col = lane & 15, rowo = (lane >> 4) * 8;

  const bf16_t* whA = wh_bf + (size_t)(rowbase + ibase) * H_DIM;
  v16bf awh[8];
#pragma unroll
  for (int k = 0; k < 8; ++k) awh[k] = load_a_frag(whA + k * 32, H_DIM, lane);

  v8f acc[4];
#pragma unroll
  for (int t = 0; t < 4; ++t) acc[t] = vzero8();

  if (w == 0)
    tdm_load_2d(lds_off(&lds_g[0][0][0]), g_bf + (size_t)rowbase * H_DIM,
                32, H_DIM, H_DIM);

  for (int step = 0; step < L_DIM / 32; ++step) {
    const int cur = step & 1;
    const int j0 = step * 32;
    if (w == 0) __builtin_amdgcn_s_wait_tensorcnt(0);
    __syncthreads();                           // (A) buf[cur] ready; old reads done
    if (w == 0 && step + 1 < L_DIM / 32)       // overlap next DMA with compute
      tdm_load_2d(lds_off(&lds_g[cur ^ 1][0][0]),
                  g_bf + (size_t)(rowbase + j0 + 32) * H_DIM,
                  32, H_DIM, H_DIM);

    {   // build g^T tile from the staged rows (LDS->reg->LDS transpose)
      const int r = tid >> 3, cc = (tid & 7) * 8;
      v8bf t = *(const v8bf*)&lds_g[cur][r][hbase + cc];
#pragma unroll
      for (int e = 0; e < 8; ++e) gt[cc + e][r] = t[e];
    }

    // recompute two 16x16 s tiles; B-fragments straight from the LDS stage
    const bf16_t* gB = &lds_g[cur][0][0];
    v8f s0 = vzero8(), s1 = vzero8();
#pragma unroll
    for (int k = 0; k < 8; ++k) {
      s0 = wmma_bf16(awh[k], load_bt_frag(gB + k * 32, H_DIM, lane), s0);
      s1 = wmma_bf16(awh[k], load_bt_frag(gB + 16 * H_DIM + k * 32, H_DIM, lane), s1);
    }

    // column-normalized probabilities (softmax over i => stats indexed by j)
    const float cm0 = cmax[rowbase + j0 + col];
    const float is0 = 1.0f / csum[rowbase + j0 + col];
    const float cm1 = cmax[rowbase + j0 + 16 + col];
    const float is1 = 1.0f / csum[rowbase + j0 + 16 + col];
#pragma unroll
    for (int v = 0; v < 8; ++v) {
      pshm[w][rowo + v][col]      = f2bf(__expf(s0[v] - cm0) * is0);
      pshm[w][rowo + v][col + 16] = f2bf(__expf(s1[v] - cm1) * is1);
    }
    __syncthreads();                           // (B) pshm + gt visible

    v16bf pa = load_a_frag(&pshm[w][0][0], 32, lane);
#pragma unroll
    for (int ht = 0; ht < 4; ++ht)
      acc[ht] = wmma_bf16(pa, load_bt_frag(&gt[ht * 16][0], 32, lane), acc[ht]);
  }

#pragma unroll
  for (int ht = 0; ht < 4; ++ht) {
#pragma unroll
    for (int v = 0; v < 8; ++v) {
      const size_t idx = (size_t)(rowbase + ibase + rowo + v) * H_DIM + hbase + ht * 16 + col;
      const float cv = acc[ht][v];
      c_bf[idx]  = f2bf(cv);
      gc_bf[idx] = f2bf(cv * g32[idx]);
    }
  }
}

// ---------------------------------------------------------------------------
// K4: out = relu([g, c, g*c] @ W_out^T + b_out); K=768 over three segments.
// ---------------------------------------------------------------------------
__global__ void out_gemm_kernel(const bf16_t* __restrict__ g_bf,
                                const bf16_t* __restrict__ c_bf,
                                const bf16_t* __restrict__ gc_bf,
                                const bf16_t* __restrict__ wo_bf,
                                const float* __restrict__ b_out,
                                float* __restrict__ out) {
  const int lane = threadIdx.x & 31;
  const int gw = blockIdx.x * 8 + (threadIdx.x >> 5);
  const int tm = gw >> 4, tn = gw & 15;
  const bf16_t* segs[3] = { g_bf, c_bf, gc_bf };
  v8f acc = vzero8();
#pragma unroll
  for (int seg = 0; seg < 3; ++seg) {
    const bf16_t* X  = segs[seg] + tm * 16 * H_DIM;
    const bf16_t* Wb = wo_bf + tn * 16 * (3 * H_DIM) + seg * H_DIM;
#pragma unroll
    for (int k = 0; k < 8; ++k)
      acc = wmma_bf16(load_a_frag(X + k * 32, H_DIM, lane),
                      load_bt_frag(Wb + k * 32, 3 * H_DIM, lane), acc);
  }
  const int col = lane & 15, rowo = (lane >> 4) * 8;
  const float bb = b_out[tn * 16 + col];
  float* D = out + tm * 16 * H_DIM + tn * 16;
#pragma unroll
  for (int v = 0; v < 8; ++v)
    D[(rowo + v) * H_DIM + col] = fmaxf(acc[v] + bb, 0.0f);
}

// ---------------------------------------------------------------------------
extern "C" void kernel_launch(void* const* d_in, const int* in_sizes, int n_in,
                              void* d_out, int out_size, void* d_ws, size_t ws_size,
                              hipStream_t stream) {
  (void)in_sizes; (void)n_in; (void)out_size; (void)ws_size;
  const float* g  = (const float*)d_in[0];   // [8,2048,256]
  const float* WP = (const float*)d_in[1];   // [256,256]
  const float* Wo = (const float*)d_in[2];   // [256,768]
  const float* bo = (const float*)d_in[3];   // [256]
  float* out = (float*)d_out;                // [8,2048,256]

  char* ws = (char*)d_ws;
  const size_t NG = (size_t)B_DIM * L_DIM * H_DIM;   // 4,194,304 elements
  bf16_t* g_bf  = (bf16_t*)(ws + 0 * NG);
  bf16_t* wh_bf = (bf16_t*)(ws + 2 * NG);
  bf16_t* c_bf  = (bf16_t*)(ws + 4 * NG);
  bf16_t* gc_bf = (bf16_t*)(ws + 6 * NG);
  bf16_t* wp_bf = (bf16_t*)(ws + 8 * NG);
  bf16_t* wo_bf = (bf16_t*)(ws + 8 * NG + 2 * H_DIM * H_DIM);
  float*  cmax  = (float*)(ws + 8 * NG + 2 * H_DIM * H_DIM + 2 * H_DIM * 3 * H_DIM);
  float*  csum  = cmax + B_DIM * L_DIM;

  cvt_f32_bf16<<<2048, 256, 0, stream>>>(g, g_bf, (int)NG);
  cvt_f32_bf16<<<64,   256, 0, stream>>>(WP, wp_bf, H_DIM * H_DIM);
  cvt_f32_bf16<<<192,  256, 0, stream>>>(Wo, wo_bf, H_DIM * 3 * H_DIM);
  wh_gemm_kernel<<<2048, 256, 0, stream>>>(g_bf, wp_bf, wh_bf);
  colstats_kernel<<<128, 256, 0, stream>>>(wh_bf, g_bf, cmax, csum);
  ct_kernel<<<512, 256, 0, stream>>>(wh_bf, g_bf, g, cmax, csum, c_bf, gc_bf);
  out_gemm_kernel<<<2048, 256, 0, stream>>>(g_bf, c_bf, gc_bf, wo_bf, bo, out);
}
